// cResNet_5789615915574
// MI455X (gfx1250) — matrix operand order repaired and checked by reading, not compiled, not run on hardware
//
#include <hip/hip_runtime.h>
#include <hip/hip_bf16.h>

// ---------------------------------------------------------------------------
// cResNet recurrence on MI455X (gfx1250), wave32 + WMMA bf16.
//
//   Y_{t+1}[b,n] = Y_t[b,n] + sum_{k,m} (dx[b,k]*Y_t[b,m]) * A[k,n,m]
//                           + sum_k     dx[b,k]            * bk[k,n]
//   out[b] = sum_n Y_50[b,n] * v[n]
//
// Per step, per 64-row block: one GEMM Z(64x4128) x AT^T(4128x256) with
// K = d*N + 32 (bias folded into the last K-block).  Y's master copy lives in
// the f32 WMMA accumulators across all 50 steps; LDS holds only a bf16
// transposed view of Y used to build the A operand Z = dx (.) Y with packed
// native bf16 multiplies.  B fragments are explicitly double-buffered
// (bbuf[2][4]) with a prefetch chain across mb, the bias block, and time
// steps.  amdgpu_num_vgpr(256) pins the allocation at the 256-VGPR window
// (live set ~230) to avoid both s_set_vgpr_msb paging and scratch spills.
// ---------------------------------------------------------------------------

typedef __bf16 bf16_t;
typedef bf16_t v16bf __attribute__((ext_vector_type(16)));
typedef bf16_t v8bf  __attribute__((ext_vector_type(8)));
typedef float  v8f   __attribute__((ext_vector_type(8)));
typedef float  v4f   __attribute__((ext_vector_type(4)));

#define NB       256          // N (feature dim)
#define DD       16           // d
#define KTOT     4128         // d*N + 32 (bias block)
#define ROWS_WG  64           // batch rows per workgroup
#define YBSTRIDE 264          // bf16 Y row stride (elems); 132 dwords = 4 mod 64
#define DXSTRIDE 20           // f32 dx row stride (floats)

// -------------------------- AT prep kernel --------------------------------
// AT[n][kk] (bf16, row-major, 256 x 4128):
//   kk <  4096 : (1/16) * Aks[k][n][m],  k=kk>>8, m=kk&255
//   kk >= 4096 : e=kk-4096;  e<16 ? bks[e][n] : 0
__global__ void prep_at_kernel(const float* __restrict__ Aks,
                               const float* __restrict__ bks,
                               bf16_t* __restrict__ AT) {
    int total = NB * KTOT;
    for (int idx = blockIdx.x * blockDim.x + threadIdx.x; idx < total;
         idx += gridDim.x * blockDim.x) {
        int n  = idx / KTOT;
        int kk = idx - n * KTOT;
        float val;
        if (kk < 4096) {
            int k = kk >> 8;
            int m = kk & 255;
            val = Aks[((k << 8) + n) * 256 + m] * 0.0625f;  // STD_A/sqrt(N)
        } else {
            int e = kk - 4096;
            val = (e < DD) ? bks[e * NB + n] : 0.0f;        // STD_B = 1
        }
        AT[idx] = (bf16_t)val;
    }
}

// -------------------------- fragment helpers ------------------------------
// B fragment (32K x 16N): lane L holds column n = L&15; elems 0-7 hold
// K = hi*8 + e, elems 8-15 hold K = 16 + hi*8 + (e-8) (AT row n, col kb*32+hi*8).
__device__ __forceinline__ v16bf load_bfrag(const bf16_t* __restrict__ p) {
    v8bf lo = *(const v8bf*)(p);
    v8bf hi = *(const v8bf*)(p + 16);
    return __builtin_shufflevector(lo, hi, 0, 1, 2, 3, 4, 5, 6, 7,
                                   8, 9, 10, 11, 12, 13, 14, 15);
}

// A fragment (16M x 32K): lane L holds row r = L&15;
//   hi=0: elems 0-7 <-> K 0-7,  elems 8-15 <-> K 16-23
//   hi=1: elems 0-7 <-> K 8-15, elems 8-15 <-> K 24-31
__device__ __forceinline__ v16bf load_y16(const bf16_t* ybrow, int m0,
                                          int hi8) {
    v8bf lo = *(const v8bf*)(ybrow + m0 + hi8);
    v8bf hi = *(const v8bf*)(ybrow + m0 + 16 + hi8);
    return __builtin_shufflevector(lo, hi, 0, 1, 2, 3, 4, 5, 6, 7,
                                   8, 9, 10, 11, 12, 13, 14, 15);
}

// Broadcast one bf16 across a v16bf (lowers to op_sel operand broadcast).
__device__ __forceinline__ v16bf splat16(bf16_t s) {
    v16bf v;
#pragma unroll
    for (int i = 0; i < 16; ++i) v[i] = s;
    return v;
}

// Bias K-block: Z[row, K] = dx[row, K] for K<16 else 0 (from f32 dx in LDS).
__device__ __forceinline__ v16bf build_zbias(const float* dxrow, int hi8) {
    const v4f a0 = *(const v4f*)(dxrow + hi8);
    const v4f a1 = *(const v4f*)(dxrow + hi8 + 4);
    v16bf z;
#pragma unroll
    for (int i = 0; i < 4; ++i) {
        z[i]      = (bf16_t)a0[i];
        z[4 + i]  = (bf16_t)a1[i];
        z[8 + i]  = (bf16_t)0.0f;
        z[12 + i] = (bf16_t)0.0f;
    }
    return z;
}

#define WMMA_BF16(a, b, c)                                                    \
    __builtin_amdgcn_wmma_f32_16x16x32_bf16(false, (a), false, (b),           \
                                            (short)0, (c), false, false)

// -------------------------- main kernel -----------------------------------
__global__ __launch_bounds__(256)
__attribute__((amdgpu_num_vgpr(256))) void cresnet_kernel(
    const float* __restrict__ X,        // (16384, 51, 16)
    const float* __restrict__ vparam,   // (256,)
    const float* __restrict__ z0param,  // (256,)
    const bf16_t* __restrict__ AT,      // (256, 4128) bf16
    float* __restrict__ out) {          // (16384,)
    __shared__ bf16_t Yb[ROWS_WG * YBSTRIDE];     // 33.8 KB bf16 view of Y
    __shared__ float  dxlds[ROWS_WG * DXSTRIDE];  // 5 KB
    __shared__ float  outred[ROWS_WG];

    const int tid  = threadIdx.x;
    const int w    = tid >> 5;        // wave 0..7
    const int lane = tid & 31;
    const int r    = lane & 15;
    const int hi   = lane >> 4;
    const int hi8  = hi * 8;
    const int mh   = w & 1;           // M half: rows 0-31 / 32-63
    const int nq   = w >> 1;          // N quarter: 64 cols each
    const int rowbase = blockIdx.x * ROWS_WG;

    // --- init accumulators (Y0 fragments) and bf16 LDS mirror of Y0 ---
    v8f acc[2][4];
    float z0n[4];
#pragma unroll
    for (int nt = 0; nt < 4; ++nt) z0n[nt] = z0param[nq * 64 + nt * 16 + r];
#pragma unroll
    for (int mt = 0; mt < 2; ++mt)
#pragma unroll
        for (int nt = 0; nt < 4; ++nt)
#pragma unroll
            for (int j = 0; j < 8; ++j) acc[mt][nt][j] = z0n[nt];

    for (int idx = tid; idx < ROWS_WG * NB; idx += 256) {
        int row = idx >> 8;
        int m   = idx & 255;
        Yb[row * YBSTRIDE + m] = (bf16_t)z0param[m];
    }
    if (tid < ROWS_WG) outred[tid] = 0.0f;

    // --- per-lane AT base pointer for this lane's B columns ---
    // atbase + nt*16*KTOT + kb*32 + hi*8  (nt offsets fold into immediates)
    const bf16_t* atbase = AT + (size_t)(nq * 64 + r) * KTOT + hi * 8;

    // --- incremental dX: thread (row64, quad) owns 4 of 16 components ---
    const int row64 = tid >> 2;
    const int quad  = tid & 3;
    const float* Xrow = X + (size_t)(rowbase + row64) * 816 + quad * 4;
    v4f xprev = *(const v4f*)(Xrow);

    const bf16_t* ybrow0 = &Yb[(mh * 32 + r) * YBSTRIDE];
    const bf16_t* ybrow1 = &Yb[(mh * 32 + 16 + r) * YBSTRIDE];
    const float* dxrow0  = &dxlds[(mh * 32 + r) * DXSTRIDE];
    const float* dxrow1  = &dxlds[(mh * 32 + 16 + r) * DXSTRIDE];

    // --- B-fragment double buffer; prime with kb = 0 ---
    v16bf bbuf[2][4];
#pragma unroll
    for (int nt = 0; nt < 4; ++nt)
        bbuf[0][nt] = load_bfrag(atbase + (size_t)nt * 16 * KTOT);

    // ============================ 50 Euler steps ============================
    for (int t = 0; t < 50; ++t) {
        // dx_t = X[:, t+1, :] - X[:, t, :]
        v4f xcur = *(const v4f*)(Xrow + (size_t)(t + 1) * 16);
        v4f dx4  = xcur - xprev;
        xprev = xcur;
        *(v4f*)&dxlds[row64 * DXSTRIDE + quad * 4] = dx4;
        __syncthreads();  // dxlds + previous-step Yb visible

        // --- preconvert this lane's 16 dx components to bf16 (registers) ---
        bf16_t dxb0[DD], dxb1[DD];
#pragma unroll
        for (int q = 0; q < 4; ++q) {
            v4f a = *(const v4f*)(dxrow0 + 4 * q);
            v4f b = *(const v4f*)(dxrow1 + 4 * q);
#pragma unroll
            for (int i = 0; i < 4; ++i) {
                dxb0[4 * q + i] = (bf16_t)a[i];
                dxb1[4 * q + i] = (bf16_t)b[i];
            }
        }

        // --- main GEMM over K = 4096 (+32 bias).  mb rolled, kk unrolled;
        //     B double-buffered: group kk+1 loads while group kk computes. ---
#pragma unroll 1
        for (int mb = 0; mb < 8; ++mb) {
            const int m0 = mb * 32;
            const v16bf y0 = load_y16(ybrow0, m0, hi8);
            const v16bf y1 = load_y16(ybrow1, m0, hi8);
#pragma unroll
            for (int kk = 0; kk < DD; ++kk) {
                const int cur = kk & 1;
                const int nxt = cur ^ 1;
                // prefetch the next K-block: next kk, else next mb's kk=0,
                // else (mb==7) the bias block kb=128.
                const int kbn = (kk < DD - 1) ? ((kk + 1) * 8 + mb)
                                              : ((mb < 7) ? (mb + 1) : 128);
#pragma unroll
                for (int nt = 0; nt < 4; ++nt)
                    bbuf[nxt][nt] = load_bfrag(
                        atbase + (size_t)nt * 16 * KTOT + (size_t)kbn * 32);
                // Z = dx * Y in native packed bf16 (v_pk_mul_bf16)
                v16bf za = y0 * splat16(dxb0[kk]);
                v16bf zb = y1 * splat16(dxb1[kk]);
#pragma unroll
                for (int nt = 0; nt < 4; ++nt) {
                    acc[0][nt] = WMMA_BF16(za, bbuf[cur][nt], acc[0][nt]);
                    acc[1][nt] = WMMA_BF16(zb, bbuf[cur][nt], acc[1][nt]);
                }
            }
        }
        // --- bias K-block (kb = 128, parity 0): Z = dx, B = bks columns ---
        {
            v16bf za = build_zbias(dxrow0, hi8);
            v16bf zb = build_zbias(dxrow1, hi8);
#pragma unroll
            for (int nt = 0; nt < 4; ++nt) {
                acc[0][nt] = WMMA_BF16(za, bbuf[0][nt], acc[0][nt]);
                acc[1][nt] = WMMA_BF16(zb, bbuf[0][nt], acc[1][nt]);
            }
            // prefetch kb = 0 for the next step (overlaps writeback+barrier)
#pragma unroll
            for (int nt = 0; nt < 4; ++nt)
                bbuf[0][nt] = load_bfrag(atbase + (size_t)nt * 16 * KTOT);
        }
        __syncthreads();  // everyone done reading Yb of this step

        // --- write updated Y fragments back to LDS as bf16 (transposed) ---
#pragma unroll
        for (int mt = 0; mt < 2; ++mt)
#pragma unroll
            for (int j = 0; j < 8; ++j) {
                const int row = mh * 32 + mt * 16 + j + hi8;
#pragma unroll
                for (int nt = 0; nt < 4; ++nt)
                    Yb[row * YBSTRIDE + nq * 64 + nt * 16 + r] =
                        (bf16_t)acc[mt][nt][j];
            }
        // next iteration's first __syncthreads() orders these stores
    }

    // ============================ out = Y @ v ===============================
    float vv[4];
#pragma unroll
    for (int nt = 0; nt < 4; ++nt)
        vv[nt] = vparam[nq * 64 + nt * 16 + r] * 0.0625f;  // 1/sqrt(N)

    __syncthreads();
#pragma unroll
    for (int mt = 0; mt < 2; ++mt)
#pragma unroll
        for (int j = 0; j < 8; ++j) {
            float p = 0.0f;
#pragma unroll
            for (int nt = 0; nt < 4; ++nt) p += acc[mt][nt][j] * vv[nt];
            atomicAdd(&outred[mh * 32 + mt * 16 + j + hi8], p);
        }
    __syncthreads();
    if (tid < ROWS_WG) out[rowbase + tid] = outred[tid];
}

// -------------------------- launcher --------------------------------------
extern "C" void kernel_launch(void* const* d_in, const int* in_sizes, int n_in,
                              void* d_out, int out_size, void* d_ws,
                              size_t ws_size, hipStream_t stream) {
    const float* X    = (const float*)d_in[0];  // (16384, 51, 16)
    const float* v    = (const float*)d_in[1];  // (256,)
    const float* z0   = (const float*)d_in[2];  // (256,)
    const float* Aks  = (const float*)d_in[3];  // (16, 256, 256)
    const float* bks  = (const float*)d_in[4];  // (16, 256)
    float* out        = (float*)d_out;          // (16384,)

    bf16_t* AT = (bf16_t*)d_ws;                 // 256*4128*2 B ~ 2.02 MB

    prep_at_kernel<<<1024, 256, 0, stream>>>(Aks, bks, AT);
    cresnet_kernel<<<16384 / ROWS_WG, 256, 0, stream>>>(X, v, z0, AT, out);
}
